// MSMambaLayer_9749575762030
// MI455X (gfx1250) — compile-verified, matrix operand order checked by reading, not run on hardware
//
#include <hip/hip_runtime.h>
#include <cmath>
#include <cstdint>

// ---------------------------------------------------------------------------
// MSMambaLayer forward for MI455X (gfx1250). All GEMM-shaped ops run through a
// bf16 WMMA kernel (v_wmma_f32_16x16x32_bf16) with the weight tile staged in
// LDS via async-to-LDS copies; everything else is fp32 VALU.
// ---------------------------------------------------------------------------

typedef unsigned short u16;
typedef unsigned int   u32;
typedef __attribute__((ext_vector_type(16))) __bf16 v16bf;
typedef __attribute__((ext_vector_type(8)))  float  v8f;
typedef __attribute__((ext_vector_type(8)))  u32    v8u;
typedef __attribute__((ext_vector_type(4)))  int    v4i;

// Pointer-to-qualified-pointee typedefs for the async-to-LDS builtin:
// arg0 is an AS1 (global) int4*, arg1 an AS3 (LDS) int4*.
typedef v4i __attribute__((address_space(1)))* as1_v4i;
typedef v4i __attribute__((address_space(3)))* as3_v4i;

static constexpr int cB   = 4;
static constexpr int cCIN = 128;
static constexpr int cHW  = 4096;   // 64*64
static constexpr int cL   = 4096;
static constexpr int cM   = cB * cHW;  // 16384 rows
static constexpr float cEPS = 1e-5f;

__device__ __forceinline__ u16 f2bf(float f) {
  u32 u = __builtin_bit_cast(u32, f);
  u32 r = u + 0x7fffu + ((u >> 16) & 1u);
  return (u16)(r >> 16);
}
__device__ __forceinline__ float siluf(float x) { return x / (1.f + __expf(-x)); }
__device__ __forceinline__ float softplusf(float x) { return x > 20.f ? x : log1pf(__expf(x)); }

// ---------------------------------------------------------------------------
// Generic bf16 WMMA GEMM:  C[m,n] = act( sum_k A[m,k]*W[n,k] + bias[n] )
//                                   (+ rscale * resid[m,n])
// A: (cM x K) bf16 row-major, lda (elements). W: (Npad x K) bf16 row-major,
// zero padded rows/cols. Block = 4 waves on 4 consecutive M-tiles sharing one
// N-tile; the 16xK weight tile is staged once into LDS (async-to-LDS when the
// toolchain exposes the builtin) with rows padded to K+8 halfwords so the
// ds_load_b128 fragment reads rotate across LDS banks.
// Fragment layout per CDNA5 ISA 7.12.2 (16-bit A 16x32): lane r=l&15 holds
// row/col r; hi half-wave takes K+8; pairs (K0..7,K16..23)/(K8..15,K24..31).
// ---------------------------------------------------------------------------
__global__ __launch_bounds__(128)
void gemm_bf16_wmma(const u16* __restrict__ A, int lda,
                    const u16* __restrict__ W, int K,
                    const float* __restrict__ bias,
                    int Nreal, int act,
                    const float* __restrict__ resid, int ldr,
                    const float* __restrict__ rscale,
                    float* __restrict__ Cf, u16* __restrict__ Cb,
                    int ldc, int ncol0)
{
  __shared__ u16 ldsW[16 * 232];        // 16 rows x (Kmax=224 + 8 pad)
  const int ldl = K + 8;                // padded LDS row stride (halfwords)

  // --- stage the 16xK weight tile into LDS (per-lane 16B async copies) ---
  {
    const int tid = threadIdx.y * 32 + threadIdx.x;
    const int cpr = K >> 3;             // 16-byte chunks per row
    const int nchunks = 16 * cpr;
    for (int c = tid; c < nchunks; c += 128) {
      int row = c / cpr;
      int kc  = (c - row * cpr) * 8;
      const u16* src = W + (size_t)(blockIdx.x * 16 + row) * K + kc;
      u16*       dst = &ldsW[row * ldl + kc];
#if __has_builtin(__builtin_amdgcn_global_load_async_to_lds_b128)
      __builtin_amdgcn_global_load_async_to_lds_b128(
          (as1_v4i)(uintptr_t)src,
          (as3_v4i)(u32)(uintptr_t)dst,
          0, 0);
#else
      *(uint4*)dst = *(const uint4*)src;   // fallback: vmem load + ds_store_b128
#endif
    }
#if __has_builtin(__builtin_amdgcn_global_load_async_to_lds_b128)
#if __has_builtin(__builtin_amdgcn_s_wait_asynccnt)
    __builtin_amdgcn_s_wait_asynccnt(0);
#else
    asm volatile("s_wait_asynccnt 0" ::: "memory");
#endif
#endif
    __syncthreads();                    // reconverged: EXEC all-ones for WMMA
  }

  const int lane = threadIdx.x;
  const int r    = lane & 15;
  const int hi   = lane >> 4;
  const int mt   = blockIdx.y * 4 + threadIdx.y;
  const int nt   = blockIdx.x;

  const u16* Arow = A + (size_t)(mt * 16 + r) * lda + hi * 8;
  const u16* Brow = &ldsW[r * ldl + hi * 8];

  v8f acc = {};
  for (int k = 0; k < K; k += 32) {
    uint4 a0 = *(const uint4*)(Arow + k);        // global_load_b128 (clause)
    uint4 a1 = *(const uint4*)(Arow + k + 16);
    uint4 b0 = *(const uint4*)(Brow + k);        // ds_load_b128
    uint4 b1 = *(const uint4*)(Brow + k + 16);
    if (k + 32 < K) {                            // uniform; global_prefetch_b8
      __builtin_prefetch(Arow + k + 32, 0, 1);
    }
    v8u av, bv;
    av[0]=a0.x; av[1]=a0.y; av[2]=a0.z; av[3]=a0.w;
    av[4]=a1.x; av[5]=a1.y; av[6]=a1.z; av[7]=a1.w;
    bv[0]=b0.x; bv[1]=b0.y; bv[2]=b0.z; bv[3]=b0.w;
    bv[4]=b1.x; bv[5]=b1.y; bv[6]=b1.z; bv[7]=b1.w;
    acc = __builtin_amdgcn_wmma_f32_16x16x32_bf16(
        false, __builtin_bit_cast(v16bf, av),
        false, __builtin_bit_cast(v16bf, bv),
        (short)0, acc, false, false);
  }

  const int n = nt * 16 + r;
  if (n < Nreal) {
    float bn = bias ? bias[n] : 0.f;
    float rs = rscale ? rscale[0] : 1.f;
#pragma unroll
    for (int i = 0; i < 8; ++i) {
      int m = mt * 16 + hi * 8 + i;     // C layout: VGPR i -> M = i + 8*hi
      float v = acc[i] + bn;
      if (act == 1) v = fmaxf(v, 0.f);
      if (resid) v += rs * resid[(size_t)m * ldr + n];
      if (Cf) Cf[(size_t)m * ldc + ncol0 + n] = v;
      if (Cb) Cb[(size_t)m * ldc + ncol0 + n] = f2bf(v);
    }
  }
}

// ---------------------------------------------------------------------------
// Weight prep
// ---------------------------------------------------------------------------
__global__ void k_foldbn(const float* __restrict__ w, const float* __restrict__ gamma,
                         const float* __restrict__ beta, const float* __restrict__ mean,
                         const float* __restrict__ var, int K, int Kpad,
                         u16* __restrict__ wb, float* __restrict__ bias, int n_tot)
{
  int idx = blockIdx.x * blockDim.x + threadIdx.x;
  if (idx >= n_tot) return;
  int n = idx / Kpad, kp = idx % Kpad;
  float scale = gamma[n] * rsqrtf(var[n] + cEPS);
  wb[idx] = (kp < K) ? f2bf(w[(size_t)n * K + kp] * scale) : (u16)0;
  if (kp == 0) bias[n] = beta[n] - mean[n] * scale;
}

__global__ void k_cvtw(const float* __restrict__ w, int Nreal, int K,
                       u16* __restrict__ wb, int n_tot)
{
  int idx = blockIdx.x * blockDim.x + threadIdx.x;
  if (idx >= n_tot) return;
  int n = idx / K, k = idx % K;
  wb[idx] = (n < Nreal) ? f2bf(w[(size_t)n * K + k]) : (u16)0;
}

// ---------------------------------------------------------------------------
// Layout staging
// ---------------------------------------------------------------------------
__global__ void k_nchw_to_row(const float* __restrict__ X, u16* __restrict__ Y, int n_tot)
{
  int idx = blockIdx.x * blockDim.x + threadIdx.x;   // idx = m*128 + c
  if (idx >= n_tot) return;
  int m = idx >> 7, c = idx & 127;
  int b = m >> 12, hw = m & 4095;
  Y[idx] = f2bf(X[((size_t)(b * cCIN + c)) * cHW + hw]);
}

__global__ void k_maxpool3_row(const float* __restrict__ X, u16* __restrict__ Y, int n_tot)
{
  int idx = blockIdx.x * blockDim.x + threadIdx.x;   // idx = m*128 + c
  if (idx >= n_tot) return;
  int m = idx >> 7, c = idx & 127;
  int b = m >> 12, hw = m & 4095, y = hw >> 6, x = hw & 63;
  const float* base = X + ((size_t)(b * cCIN + c)) * cHW;
  float mx = -3.4e38f;
  for (int dy = -1; dy <= 1; ++dy)
    for (int dx = -1; dx <= 1; ++dx) {
      int iy = y + dy, ix = x + dx;
      if (iy >= 0 && iy < 64 && ix >= 0 && ix < 64)
        mx = fmaxf(mx, base[iy * 64 + ix]);
    }
  Y[idx] = f2bf(mx);
}

__global__ void k_im2col(const u16* __restrict__ In, int Cin, int ks, int pad,
                         int Kpad, u16* __restrict__ Out, int n_tot)
{
  int idx = blockIdx.x * blockDim.x + threadIdx.x;   // idx = m*Kpad + kk
  if (idx >= n_tot) return;
  int m = idx / Kpad, kk = idx % Kpad;
  int K = Cin * ks * ks;
  u16 v = 0;
  if (kk < K) {
    int ci = kk / (ks * ks), rem = kk % (ks * ks);
    int ky = rem / ks, kx = rem % ks;
    int b = m >> 12, hw = m & 4095, y = hw >> 6, x = hw & 63;
    int iy = y + ky - pad, ix = x + kx - pad;
    if (iy >= 0 && iy < 64 && ix >= 0 && ix < 64)
      v = In[(size_t)((b << 12) + iy * 64 + ix) * Cin + ci];
  }
  Out[idx] = v;
}

// ---------------------------------------------------------------------------
// LayerNorm over last dim C (fp32 in, fp32 + bf16 out)
// ---------------------------------------------------------------------------
__global__ void k_layernorm(const float* __restrict__ X, int C,
                            const float* __restrict__ g, const float* __restrict__ bb,
                            float* __restrict__ Yf, u16* __restrict__ Yb, int Mrows)
{
  int m = blockIdx.x * blockDim.x + threadIdx.x;
  if (m >= Mrows) return;
  const float* x = X + (size_t)m * C;
  float mu = 0.f;
  for (int c = 0; c < C; ++c) mu += x[c];
  mu /= C;
  float var = 0.f;
  for (int c = 0; c < C; ++c) { float d = x[c] - mu; var += d * d; }
  var /= C;
  float inv = rsqrtf(var + cEPS);
  for (int c = 0; c < C; ++c) {
    float v = (x[c] - mu) * inv * g[c] + bb[c];
    if (Yf) Yf[(size_t)m * C + c] = v;
    if (Yb) Yb[(size_t)m * C + c] = f2bf(v);
  }
}

// ---------------------------------------------------------------------------
// Causal depthwise conv (width 4) + SiLU.  XZ: (M,128), xc = cols [0,64)
// ---------------------------------------------------------------------------
__global__ void k_dwconv_silu(const float* __restrict__ XZ,
                              const float* __restrict__ cw, const float* __restrict__ cb,
                              float* __restrict__ Xf, u16* __restrict__ Xb, int n_tot)
{
  int idx = blockIdx.x * blockDim.x + threadIdx.x;   // idx = m*64 + d
  if (idx >= n_tot) return;
  int m = idx >> 6, d = idx & 63;
  int l = m & (cL - 1);
  float s = cb[d];
#pragma unroll
  for (int j = 0; j < 4; ++j) {
    int ll = l - 3 + j;
    if (ll >= 0) s += cw[d * 4 + j] * XZ[(size_t)(m - 3 + j) * 128 + d];
  }
  s = siluf(s);
  Xf[idx] = s;
  Xb[idx] = f2bf(s);
}

// dt = softplus(x_dbl[:, :2] @ dt_proj_w^T + dt_bias)
__global__ void k_dt_softplus(const float* __restrict__ XD, const float* __restrict__ W,
                              const float* __restrict__ bias, float* __restrict__ DT, int n_tot)
{
  int idx = blockIdx.x * blockDim.x + threadIdx.x;   // idx = m*64 + d
  if (idx >= n_tot) return;
  int m = idx >> 6, d = idx & 63;
  float v = XD[(size_t)m * 48 + 0] * W[d * 2 + 0]
          + XD[(size_t)m * 48 + 1] * W[d * 2 + 1] + bias[d];
  DT[idx] = softplusf(v);
}

// ---------------------------------------------------------------------------
// Selective scan. One block (1024 threads) per batch sequence: thread =
// (d_inner, d_state) pair; state contraction via width-16 shuffle tree.
// x_dbl layout: [0,2)=dt_rank, [2,18)=B, [18,34)=C (padded to 48).
// ---------------------------------------------------------------------------
__global__ __launch_bounds__(1024)
void k_ssm_scan(const float* __restrict__ Alog, const float* __restrict__ Dp,
                const float* __restrict__ XZ, const float* __restrict__ XD,
                const float* __restrict__ DT, const float* __restrict__ XC,
                u16* __restrict__ Yb)
{
  int b = blockIdx.x;
  int d = threadIdx.x >> 4;
  int s = threadIdx.x & 15;
  float Ads = -__expf(Alog[d * 16 + s]);
  float Dd  = Dp[d];
  float h   = 0.f;
  size_t base = (size_t)b * cL;
  for (int l = 0; l < cL; ++l) {
    size_t m  = base + l;
    float dt  = DT[m * 64 + d];
    float xc  = XC[m * 64 + d];
    float Bv  = XD[m * 48 + 2 + s];
    float Cv  = XD[m * 48 + 18 + s];
    h = __expf(dt * Ads) * h + (dt * xc) * Bv;
    float p = h * Cv;
    p += __shfl_xor(p, 8, 16);
    p += __shfl_xor(p, 4, 16);
    p += __shfl_xor(p, 2, 16);
    p += __shfl_xor(p, 1, 16);
    if (s == 0) {
      float z = XZ[m * 128 + 64 + d];
      float y = (p + xc * Dd) * siluf(z);
      Yb[m * 64 + d] = f2bf(y);
    }
  }
}

// out_nchw = x + gemm_out(row-major (M,128)) transposed back
__global__ void k_residual_out(const float* __restrict__ X, const float* __restrict__ G,
                               float* __restrict__ O, int n_tot)
{
  int idx = blockIdx.x * blockDim.x + threadIdx.x;   // NCHW flat
  if (idx >= n_tot) return;
  int hw = idx & 4095, c = (idx >> 12) & 127, b = idx >> 19;
  O[idx] = X[idx] + G[((size_t)(b << 12) + hw) * 128 + c];
}

// ---------------------------------------------------------------------------
// Host orchestration
// ---------------------------------------------------------------------------
static inline void launch_gemm(hipStream_t st, const u16* A, int lda, const u16* W, int K,
                               const float* bias, int Npad, int Nreal, int act,
                               const float* resid, int ldr, const float* rscale,
                               float* Cf, u16* Cb, int ldc, int ncol0)
{
  dim3 grid(Npad / 16, cM / 64);
  dim3 blk(32, 4);
  hipLaunchKernelGGL(gemm_bf16_wmma, grid, blk, 0, st,
                     A, lda, W, K, bias, Nreal, act, resid, ldr, rscale, Cf, Cb, ldc, ncol0);
}

#define EW(kname, n, ...) do { int _n = (n);                                       \
    hipLaunchKernelGGL(kname, dim3((_n + 255) / 256), dim3(256), 0, stream,        \
                       __VA_ARGS__, _n); } while (0)

extern "C" void kernel_launch(void* const* d_in, const int* in_sizes, int n_in,
                              void* d_out, int out_size, void* d_ws, size_t ws_size,
                              hipStream_t stream)
{
  (void)in_sizes; (void)n_in; (void)out_size; (void)ws_size;
  auto F = [&](int i) { return (const float*)d_in[i]; };

  // Flattened input indices (jax tree order: sorted dict keys at each level).
  // bnp{beta,gamma,mean,var,w}: b1=0, b2a=5, b2b=10, b3a=15, b3b=20, b4=25
  // ln{1..4}_{b,g}: 30..37; m{1..4}: base 38+9i with
  //   {A_log,D,conv_b,conv_w,dt_bias,dt_proj_w,in_proj_w,out_proj_w,x_proj_w}
  // norm_b=74, norm_g=75, out_proj_w=76, skip_scale=77, x=78
  const float* x_in   = F(78);
  const float* skip   = F(77);
  const float* norm_b = F(74);
  const float* norm_g = F(75);
  const float* wfin   = F(76);

  // --- workspace bump allocator (256B aligned) ---
  char* wsb = (char*)d_ws;
  size_t off = 0;
  auto alloc = [&](size_t bytes) -> void* {
    off = (off + 255) & ~(size_t)255;
    void* p = wsb + off;
    off += bytes;
    return p;
  };

  u16*   xrow   = (u16*)  alloc((size_t)cM * 128 * 2);
  u16*   pool   = (u16*)  alloc((size_t)cM * 128 * 2);
  float* xcat   = (float*)alloc((size_t)cM * 128 * 4);
  u16*   xcatln = (u16*)  alloc((size_t)cM * 128 * 2);
  float* gout   = (float*)alloc((size_t)cM * 128 * 4);
  float* feat   = (float*)alloc((size_t)cM * 32 * 4);
  float* t_f    = (float*)alloc((size_t)cM * 32 * 4);
  u16*   t_b    = (u16*)  alloc((size_t)cM * 32 * 2);
  float* xz     = (float*)alloc((size_t)cM * 128 * 4);
  float* xca_f  = (float*)alloc((size_t)cM * 64 * 4);
  u16*   xca_b  = (u16*)  alloc((size_t)cM * 64 * 2);
  float* xdbl   = (float*)alloc((size_t)cM * 48 * 4);
  float* dtb    = (float*)alloc((size_t)cM * 64 * 4);
  u16*   y_b    = (u16*)  alloc((size_t)cM * 64 * 2);
  u16*   b2a_b  = (u16*)  alloc((size_t)cM * 16 * 2);
  u16*   b3a_b  = (u16*)  alloc((size_t)cM * 8 * 2);
  u16*   col    = (u16*)  alloc((size_t)cM * 224 * 2);

  // folded conv weights (bf16) + biases
  struct ConvW { u16* w; float* b; int K, Kpad, N, Npad; };
  auto mk = [&](int N, int Npad, int K, int Kpad) {
    ConvW c; c.N = N; c.Npad = Npad; c.K = K; c.Kpad = Kpad;
    c.w = (u16*)alloc((size_t)Npad * Kpad * 2);
    c.b = (float*)alloc((size_t)Npad * 4);
    return c;
  };
  ConvW cb1  = mk(32, 32, 128, 128);
  ConvW cb2a = mk(16, 16, 128, 128);
  ConvW cb2b = mk(32, 32, 144, 160);
  ConvW cb3a = mk(8, 16, 128, 128);
  ConvW cb3b = mk(32, 32, 200, 224);
  ConvW cb4  = mk(32, 32, 128, 128);
  u16* win[4]; u16* wxp[4]; u16* wop[4];
  for (int i = 0; i < 4; ++i) {
    win[i] = (u16*)alloc((size_t)128 * 32 * 2);   // in_proj  (128x32)
    wxp[i] = (u16*)alloc((size_t)48 * 64 * 2);    // x_proj   (34->48 x 64)
    wop[i] = (u16*)alloc((size_t)32 * 64 * 2);    // out_proj (32x64)
  }
  u16* wF = (u16*)alloc((size_t)128 * 128 * 2);   // final 1x1 conv

  // --- weight prep ---
  auto fold = [&](const ConvW& c, int base) {
    EW(k_foldbn, c.N * c.Kpad, F(base + 4), F(base + 1), F(base + 0), F(base + 2),
       F(base + 3), c.K, c.Kpad, c.w, c.b);
    if (c.Npad > c.N)  // zero pad extra output rows
      EW(k_cvtw, (c.Npad - c.N) * c.Kpad, F(base + 4), 0, c.Kpad,
         c.w + (size_t)c.N * c.Kpad);
  };
  fold(cb1, 0); fold(cb2a, 5); fold(cb2b, 10); fold(cb3a, 15); fold(cb3b, 20); fold(cb4, 25);
  for (int i = 0; i < 4; ++i) {
    int mb = 38 + 9 * i;
    EW(k_cvtw, 128 * 32, F(mb + 6), 128, 32, win[i]);
    EW(k_cvtw, 48 * 64,  F(mb + 8), 34,  64, wxp[i]);
    EW(k_cvtw, 32 * 64,  F(mb + 7), 32,  64, wop[i]);
  }
  EW(k_cvtw, 128 * 128, wfin, 128, 128, wF);

  // --- staging ---
  EW(k_nchw_to_row, cM * 128, x_in, xrow);
  EW(k_maxpool3_row, cM * 128, x_in, pool);

  // --- 4 branches ---
  for (int i = 0; i < 4; ++i) {
    // branch conv -> feat (M x 32, fp32)
    if (i == 0) {
      launch_gemm(stream, xrow, 128, cb1.w, 128, cb1.b, 32, 32, 1,
                  nullptr, 0, nullptr, feat, nullptr, 32, 0);
    } else if (i == 1) {
      launch_gemm(stream, xrow, 128, cb2a.w, 128, cb2a.b, 16, 16, 1,
                  nullptr, 0, nullptr, nullptr, b2a_b, 16, 0);
      EW(k_im2col, cM * 160, b2a_b, 16, 3, 1, 160, col);
      launch_gemm(stream, col, 160, cb2b.w, 160, cb2b.b, 32, 32, 1,
                  nullptr, 0, nullptr, feat, nullptr, 32, 0);
    } else if (i == 2) {
      launch_gemm(stream, xrow, 128, cb3a.w, 128, cb3a.b, 16, 8, 1,
                  nullptr, 0, nullptr, nullptr, b3a_b, 8, 0);
      EW(k_im2col, cM * 224, b3a_b, 8, 5, 2, 224, col);
      launch_gemm(stream, col, 224, cb3b.w, 224, cb3b.b, 32, 32, 1,
                  nullptr, 0, nullptr, feat, nullptr, 32, 0);
    } else {
      launch_gemm(stream, pool, 128, cb4.w, 128, cb4.b, 32, 32, 1,
                  nullptr, 0, nullptr, feat, nullptr, 32, 0);
    }

    // LayerNorm (C=32) -> t (fp32 + bf16)
    EW(k_layernorm, cM, feat, 32, F(31 + 2 * i), F(30 + 2 * i), t_f, t_b);

    int mb = 38 + 9 * i;
    // in_proj: (M,32) @ (128,32)^T -> xz (M,128)
    launch_gemm(stream, t_b, 32, win[i], 32, nullptr, 128, 128, 0,
                nullptr, 0, nullptr, xz, nullptr, 128, 0);
    // causal depthwise conv + silu -> xc (fp32 + bf16)
    EW(k_dwconv_silu, cM * 64, xz, F(mb + 3), F(mb + 2), xca_f, xca_b);
    // x_proj: (M,64) @ (34,64)^T -> x_dbl (M,48 padded)
    launch_gemm(stream, xca_b, 64, wxp[i], 64, nullptr, 48, 34, 0,
                nullptr, 0, nullptr, xdbl, nullptr, 48, 0);
    // dt = softplus(...)
    EW(k_dt_softplus, cM * 64, xdbl, F(mb + 5), F(mb + 4), dtb);
    // selective scan + D skip + SiLU(z) gate -> y (bf16)
    hipLaunchKernelGGL(k_ssm_scan, dim3(cB), dim3(1024), 0, stream,
                       F(mb + 0), F(mb + 1), xz, xdbl, dtb, xca_f, y_b);
    // out_proj + skip*t, written into xcat columns [32i, 32i+32)
    launch_gemm(stream, y_b, 64, wop[i], 64, nullptr, 32, 32, 0,
                t_f, 32, skip, xcat, nullptr, 128, 32 * i);
  }

  // final LN (C=128) -> bf16, final 1x1 conv, residual to NCHW output
  EW(k_layernorm, cM, xcat, 128, norm_g, norm_b, (float*)nullptr, xcatln);
  launch_gemm(stream, xcatln, 128, wF, 128, nullptr, 128, 128, 0,
              nullptr, 0, nullptr, gout, nullptr, 128, 0);
  EW(k_residual_out, cB * cCIN * cHW, x_in, gout, (float*)d_out);
}
